// ROIBrainEncoder_76312978915441
// MI455X (gfx1250) — compile-verified, matrix operand order checked by reading, not run on hardware
//
#include <hip/hip_runtime.h>
#include <math.h>

typedef __attribute__((ext_vector_type(16))) _Float16 v16h;
typedef __attribute__((ext_vector_type(8)))  _Float16 v8h;
typedef __attribute__((ext_vector_type(2)))  _Float16 v2h;
typedef __attribute__((ext_vector_type(8)))  float    v8f;

#define B_ROWS   1024
#define D_DIM    128
#define RD       1024      // R*D
#define COND     768
#define NTOKCOND 12288     // 16*768
#define LDT      40        // LDS tile stride in f16 (32 + 8 pad) -> 80B rows, 16B aligned

// ---------------------------------------------------------------------------
// helpers
// ---------------------------------------------------------------------------
__device__ __forceinline__ v2h pk2(float a, float b) {
    // v_cvt_pk_f16_f32 ; builtin returns __fp16x2, bit-identical to v2h
    return __builtin_bit_cast(v2h, __builtin_amdgcn_cvt_pkrtz(a, b));
}

__device__ __forceinline__ v16h load_frag16(const _Float16* p) {
    // Two ds_load_b128: K 0..7 -> VGPR0-3, K 16..23 -> VGPR4-7 (per-lane,
    // already offset by 8*(lane>>4)). Matches ISA 16-bit A/B fragment layout.
    v8h x0 = *(const v8h*)p;
    v8h x1 = *(const v8h*)(p + 16);
    v16h r;
#pragma unroll
    for (int i = 0; i < 8; ++i) { r[i] = x0[i]; r[i + 8] = x1[i]; }
    return r;
}

// one 128x128x32 MMA step for a wave: A strip (16 rows) x 8 N-tiles.
// B fragments preloaded in groups of 4 -> clause'd ds loads, back-to-back WMMAs.
__device__ __forceinline__ void mma_step(const _Float16* At, const _Float16* Bt,
                                         int wave, int lm, int lh, v8f* acc) {
    v16h a = load_frag16(&At[(wave * 16 + lm) * LDT + 8 * lh]);
#pragma unroll
    for (int g = 0; g < 2; ++g) {
        v16h bf[4];
#pragma unroll
        for (int q = 0; q < 4; ++q)
            bf[q] = load_frag16(&Bt[((g * 4 + q) * 16 + lm) * LDT + 8 * lh]);
#pragma unroll
        for (int q = 0; q < 4; ++q)
            acc[g * 4 + q] = __builtin_amdgcn_wmma_f32_16x16x32_f16(
                false, a, false, bf[q], (short)0, acc[g * 4 + q], false, false);
    }
}

// write a 16-float A chunk (one row, 32 K-elems split across 2 threads) packed f16
__device__ __forceinline__ void storeA16(_Float16* At, int arow, int acb,
                                         const float* aR) {
    v8h u, v;
#pragma unroll
    for (int i = 0; i < 4; ++i) {
        v2h p = pk2(aR[2 * i], aR[2 * i + 1]);
        u[2 * i] = p[0]; u[2 * i + 1] = p[1];
    }
#pragma unroll
    for (int i = 0; i < 4; ++i) {
        v2h p = pk2(aR[8 + 2 * i], aR[9 + 2 * i]);
        v[2 * i] = p[0]; v[2 * i + 1] = p[1];
    }
    *(v8h*)&At[arow * LDT + acb]     = u;
    *(v8h*)&At[arow * LDT + acb + 8] = v;
}

// write 8 transposed B K-pairs as packed b32 stores: Bt[n][2kp..2kp+1]
__device__ __forceinline__ void storeB16(_Float16* Bt, int t, const float* bR) {
#pragma unroll
    for (int j = 0; j < 8; ++j) {
        int e = t + 256 * j;
        int kp = e >> 7;
        int n  = e & 127;
        *(v2h*)&Bt[n * LDT + 2 * kp] = pk2(bR[2 * j], bR[2 * j + 1]);
    }
}

struct RoiArgs {
    const float* x[8];
    const float* w[8];
    int          K[8];
};

// ---------------------------------------------------------------------------
// Kernel 1: per-ROI ragged GEMM  h1 = gelu(ln(x_r @ W1_r + b1_r))  -> f16
// grid: (rowBlocks=8, roi=8), block 256 (8 waves), double-buffered LDS
// ---------------------------------------------------------------------------
__global__ __launch_bounds__(256) void k1_roi_gemm_ln_gelu(
    RoiArgs ra, const float* __restrict__ b1, const float* __restrict__ ln1g,
    const float* __restrict__ ln1b, _Float16* __restrict__ h1)
{
    __shared__ __align__(16) _Float16 At[2][128 * LDT];
    __shared__ __align__(16) _Float16 Bt[2][128 * LDT];

    const int r        = blockIdx.y;
    const int rowBlock = blockIdx.x;
    const int K        = ra.K[r];
    const float* __restrict__ X = ra.x[r];
    const float* __restrict__ W = ra.w[r];

    const int t    = threadIdx.x;
    const int lane = t & 31;
    const int wave = t >> 5;
    const int lm   = lane & 15;
    const int lh   = lane >> 4;

    const int arow = t >> 1;
    const int acb  = (t & 1) * 16;
    const float* xr = X + (size_t)(rowBlock * 128 + arow) * K;

    v8f acc[8];
#pragma unroll
    for (int i = 0; i < 8; ++i) acc[i] = (v8f)0.0f;

    const int kSteps = (K + 31) >> 5;
    float aR[16], bR[16];

    // load + stage chunk 0
    {
        const int k0 = 0;
        if (32 <= K) {
            const float4* s4 = (const float4*)(xr + acb);
            float4 f0 = s4[0], f1 = s4[1], f2 = s4[2], f3 = s4[3];
            aR[0]=f0.x; aR[1]=f0.y; aR[2]=f0.z; aR[3]=f0.w;
            aR[4]=f1.x; aR[5]=f1.y; aR[6]=f1.z; aR[7]=f1.w;
            aR[8]=f2.x; aR[9]=f2.y; aR[10]=f2.z; aR[11]=f2.w;
            aR[12]=f3.x; aR[13]=f3.y; aR[14]=f3.z; aR[15]=f3.w;
        } else {
#pragma unroll
            for (int i = 0; i < 16; ++i) {
                int k = acb + i;
                aR[i] = (k < K) ? xr[k] : 0.0f;
            }
        }
#pragma unroll
        for (int j = 0; j < 8; ++j) {
            int e = t + 256 * j; int kp = e >> 7; int n = e & 127;
            int k = k0 + 2 * kp;
            bR[2*j]   = (k     < K) ? W[(size_t)k * 128 + n]       : 0.0f;
            bR[2*j+1] = (k + 1 < K) ? W[(size_t)k * 128 + 128 + n] : 0.0f;
        }
        storeA16(At[0], arow, acb, aR);
        storeB16(Bt[0], t, bR);
    }
    __syncthreads();

    for (int ks = 0; ks < kSteps; ++ks) {
        const int cur = ks & 1;
        const bool more = (ks + 1 < kSteps);
        if (more) {
            const int k0 = (ks + 1) << 5;
            if (k0 + 32 < K) __builtin_prefetch(W + (size_t)(k0 + 32) * 128, 0, 1);
            if (k0 + 32 <= K) {
                const float4* s4 = (const float4*)(xr + k0 + acb);
                float4 f0 = s4[0], f1 = s4[1], f2 = s4[2], f3 = s4[3];
                aR[0]=f0.x; aR[1]=f0.y; aR[2]=f0.z; aR[3]=f0.w;
                aR[4]=f1.x; aR[5]=f1.y; aR[6]=f1.z; aR[7]=f1.w;
                aR[8]=f2.x; aR[9]=f2.y; aR[10]=f2.z; aR[11]=f2.w;
                aR[12]=f3.x; aR[13]=f3.y; aR[14]=f3.z; aR[15]=f3.w;
            } else {
#pragma unroll
                for (int i = 0; i < 16; ++i) {
                    int k = k0 + acb + i;
                    aR[i] = (k < K) ? xr[k] : 0.0f;
                }
            }
#pragma unroll
            for (int j = 0; j < 8; ++j) {
                int e = t + 256 * j; int kp = e >> 7; int n = e & 127;
                int k = k0 + 2 * kp;
                bR[2*j]   = (k     < K) ? W[(size_t)k * 128 + n]       : 0.0f;
                bR[2*j+1] = (k + 1 < K) ? W[(size_t)k * 128 + 128 + n] : 0.0f;
            }
        }
        mma_step(At[cur], Bt[cur], wave, lm, lh, acc);
        if (more) {
            storeA16(At[cur ^ 1], arow, acb, aR);
            storeB16(Bt[cur ^ 1], t, bR);
        }
        __syncthreads();
    }

    // epilogue: +b1, LayerNorm over D=128 (row lives in one 16-lane half), GELU
    float bias_n[8], g_n[8], beta_n[8];
#pragma unroll
    for (int tt = 0; tt < 8; ++tt) {
        int n = r * 128 + tt * 16 + lm;
        bias_n[tt] = b1[n]; g_n[tt] = ln1g[n]; beta_n[tt] = ln1b[n];
    }
#pragma unroll
    for (int V = 0; V < 8; ++V) {
        float vals[8], s = 0.0f, ss = 0.0f;
#pragma unroll
        for (int tt = 0; tt < 8; ++tt) {
            float v = acc[tt][V] + bias_n[tt];
            vals[tt] = v; s += v; ss += v * v;
        }
#pragma unroll
        for (int m = 1; m < 16; m <<= 1) {
            s  += __shfl_xor(s,  m, 32);
            ss += __shfl_xor(ss, m, 32);
        }
        float mu  = s  * (1.0f / 128.0f);
        float var = ss * (1.0f / 128.0f) - mu * mu;
        float rs  = rsqrtf(var + 1e-5f);
        int row = rowBlock * 128 + wave * 16 + V + 8 * lh;
#pragma unroll
        for (int tt = 0; tt < 8; ++tt) {
            float v = (vals[tt] - mu) * rs * g_n[tt] + beta_n[tt];
            float g = 0.5f * v * (1.0f + erff(v * 0.70710678118654752f));
            h1[(size_t)row * RD + r * 128 + tt * 16 + lm] = (_Float16)g;
        }
    }
}

// ---------------------------------------------------------------------------
// Kernel 2: per-ROI second linear  h2[:, r] = h1[:, r] @ W2[r] + b2[r]  (f16)
// ---------------------------------------------------------------------------
__global__ __launch_bounds__(256) void k2_roi_linear(
    const _Float16* __restrict__ h1, const float* __restrict__ W2,
    const float* __restrict__ b2, _Float16* __restrict__ h2)
{
    __shared__ __align__(16) _Float16 At[2][128 * LDT];
    __shared__ __align__(16) _Float16 Bt[2][128 * LDT];

    const int r        = blockIdx.y;
    const int rowBlock = blockIdx.x;
    const int t    = threadIdx.x;
    const int lane = t & 31;
    const int wave = t >> 5;
    const int lm   = lane & 15;
    const int lh   = lane >> 4;

    const int arow = t >> 1;
    const int acb  = (t & 1) * 16;
    const _Float16* asrc = h1 + (size_t)(rowBlock * 128 + arow) * RD + r * 128 + acb;

    v8f acc[8];
#pragma unroll
    for (int i = 0; i < 8; ++i) acc[i] = (v8f)0.0f;

    v8h a0, a1;
    float bR[16];
    // stage chunk 0
    a0 = *(const v8h*)asrc; a1 = *(const v8h*)(asrc + 8);
#pragma unroll
    for (int j = 0; j < 8; ++j) {
        int e = t + 256 * j; int kp = e >> 7; int n = e & 127;
        const float* wsrc = W2 + ((size_t)r * 128 + 2 * kp) * 128 + n;
        bR[2*j] = wsrc[0]; bR[2*j+1] = wsrc[128];
    }
    *(v8h*)&At[0][arow * LDT + acb]     = a0;
    *(v8h*)&At[0][arow * LDT + acb + 8] = a1;
    storeB16(Bt[0], t, bR);
    __syncthreads();

#pragma unroll
    for (int ks = 0; ks < 4; ++ks) {
        const int cur = ks & 1;
        const bool more = (ks + 1 < 4);
        if (more) {
            const int k0 = (ks + 1) << 5;
            a0 = *(const v8h*)(asrc + k0);
            a1 = *(const v8h*)(asrc + k0 + 8);
#pragma unroll
            for (int j = 0; j < 8; ++j) {
                int e = t + 256 * j; int kp = e >> 7; int n = e & 127;
                const float* wsrc = W2 + ((size_t)r * 128 + k0 + 2 * kp) * 128 + n;
                bR[2*j] = wsrc[0]; bR[2*j+1] = wsrc[128];
            }
        }
        mma_step(At[cur], Bt[cur], wave, lm, lh, acc);
        if (more) {
            *(v8h*)&At[cur ^ 1][arow * LDT + acb]     = a0;
            *(v8h*)&At[cur ^ 1][arow * LDT + acb + 8] = a1;
            storeB16(Bt[cur ^ 1], t, bR);
        }
        __syncthreads();
    }

#pragma unroll
    for (int tt = 0; tt < 8; ++tt) {
        int n = r * 128 + tt * 16 + lm;
        float bias = b2[n];
#pragma unroll
        for (int V = 0; V < 8; ++V) {
            int row = rowBlock * 128 + wave * 16 + V + 8 * lh;
            h2[(size_t)row * RD + n] = (_Float16)(acc[tt][V] + bias);
        }
    }
}

// ---------------------------------------------------------------------------
// Kernel 3: fused head GEMM  [1024x1024] @ [1024x(768|12288)] + bias -> fp32
// grid: (colBlocks=102, rowBlocks=8). 768 = 6*128, so each block is pure Wg or Wt
// ---------------------------------------------------------------------------
__global__ __launch_bounds__(256) void k3_head_gemm(
    const _Float16* __restrict__ h2, const float* __restrict__ Wg,
    const float* __restrict__ Wt, const float* __restrict__ bg,
    const float* __restrict__ bt, float* __restrict__ outG,
    float* __restrict__ outT)
{
    __shared__ __align__(16) _Float16 At[2][128 * LDT];
    __shared__ __align__(16) _Float16 Bt[2][128 * LDT];

    const int cBase    = blockIdx.x * 128;
    const int rowBlock = blockIdx.y;
    const bool isG     = (cBase < COND);
    const float* __restrict__ Wsrc = isG ? Wg : (Wt - COND);  // index by absolute col
    const int ldw = isG ? COND : NTOKCOND;

    const int t    = threadIdx.x;
    const int lane = t & 31;
    const int wave = t >> 5;
    const int lm   = lane & 15;
    const int lh   = lane >> 4;

    const int arow = t >> 1;
    const int acb  = (t & 1) * 16;
    const _Float16* asrc = h2 + (size_t)(rowBlock * 128 + arow) * RD + acb;

    v8f acc[8];
#pragma unroll
    for (int i = 0; i < 8; ++i) acc[i] = (v8f)0.0f;

    v8h a0, a1;
    float bR[16];
    a0 = *(const v8h*)asrc; a1 = *(const v8h*)(asrc + 8);
#pragma unroll
    for (int j = 0; j < 8; ++j) {
        int e = t + 256 * j; int kp = e >> 7; int n = e & 127;
        const float* wsrc = Wsrc + (size_t)(2 * kp) * ldw + cBase + n;
        bR[2*j] = wsrc[0]; bR[2*j+1] = wsrc[ldw];
    }
    *(v8h*)&At[0][arow * LDT + acb]     = a0;
    *(v8h*)&At[0][arow * LDT + acb + 8] = a1;
    storeB16(Bt[0], t, bR);
    __syncthreads();

    for (int ks = 0; ks < 32; ++ks) {
        const int cur = ks & 1;
        const bool more = (ks + 1 < 32);
        if (more) {
            const int k0 = (ks + 1) << 5;
            if (ks + 2 < 32)
                __builtin_prefetch(Wsrc + (size_t)(k0 + 32) * ldw + cBase, 0, 1);
            a0 = *(const v8h*)(asrc + k0);
            a1 = *(const v8h*)(asrc + k0 + 8);
#pragma unroll
            for (int j = 0; j < 8; ++j) {
                int e = t + 256 * j; int kp = e >> 7; int n = e & 127;
                const float* wsrc = Wsrc + (size_t)(k0 + 2 * kp) * ldw + cBase + n;
                bR[2*j] = wsrc[0]; bR[2*j+1] = wsrc[ldw];
            }
        }
        mma_step(At[cur], Bt[cur], wave, lm, lh, acc);
        if (more) {
            *(v8h*)&At[cur ^ 1][arow * LDT + acb]     = a0;
            *(v8h*)&At[cur ^ 1][arow * LDT + acb + 8] = a1;
            storeB16(Bt[cur ^ 1], t, bR);
        }
        __syncthreads();
    }

#pragma unroll
    for (int tt = 0; tt < 8; ++tt) {
        int col = cBase + tt * 16 + lm;
        float bias = isG ? bg[col] : bt[col - COND];
#pragma unroll
        for (int V = 0; V < 8; ++V) {
            int row = rowBlock * 128 + wave * 16 + V + 8 * lh;
            float v = acc[tt][V] + bias;
            if (isG) outG[(size_t)row * COND + col] = v;
            else     outT[(size_t)row * NTOKCOND + (col - COND)] = v;
        }
    }
}

// ---------------------------------------------------------------------------
// Kernel 4: row LayerNorm over the 768-wide global head (in place)
// ---------------------------------------------------------------------------
__global__ __launch_bounds__(256) void k4_row_ln(
    float* __restrict__ outG, const float* __restrict__ g,
    const float* __restrict__ b)
{
    const int row  = blockIdx.x;
    const int t    = threadIdx.x;
    const int lane = t & 31;
    const int wave = t >> 5;
    float* rp = outG + (size_t)row * COND;

    float v[3], s = 0.0f, ss = 0.0f;
#pragma unroll
    for (int i = 0; i < 3; ++i) {
        v[i] = rp[t + 256 * i];
        s += v[i]; ss += v[i] * v[i];
    }
#pragma unroll
    for (int m = 1; m < 32; m <<= 1) {
        s  += __shfl_xor(s,  m, 32);
        ss += __shfl_xor(ss, m, 32);
    }
    __shared__ float sb[8], ssb[8];
    __shared__ float mu_s, rs_s;
    if (lane == 0) { sb[wave] = s; ssb[wave] = ss; }
    __syncthreads();
    if (t == 0) {
        float S = 0.0f, SS = 0.0f;
#pragma unroll
        for (int w = 0; w < 8; ++w) { S += sb[w]; SS += ssb[w]; }
        float mu  = S * (1.0f / 768.0f);
        float var = SS * (1.0f / 768.0f) - mu * mu;
        mu_s = mu;
        rs_s = rsqrtf(var + 1e-5f);
    }
    __syncthreads();
    float mu = mu_s, rs = rs_s;
#pragma unroll
    for (int i = 0; i < 3; ++i) {
        int idx = t + 256 * i;
        rp[idx] = (v[i] - mu) * rs * g[idx] + b[idx];
    }
}

// ---------------------------------------------------------------------------
extern "C" void kernel_launch(void* const* d_in, const int* in_sizes, int n_in,
                              void* d_out, int out_size, void* d_ws, size_t ws_size,
                              hipStream_t stream)
{
    (void)in_sizes; (void)n_in; (void)out_size; (void)ws_size;

    RoiArgs ra;
    static const int Ks[8] = {1200, 800, 2500, 900, 700, 4000, 3500, 2800};
    for (int i = 0; i < 8; ++i) {
        ra.x[i] = (const float*)d_in[i];
        ra.w[i] = (const float*)d_in[8 + i];
        ra.K[i] = Ks[i];
    }
    const float* b1    = (const float*)d_in[16];
    const float* ln1g  = (const float*)d_in[17];
    const float* ln1b  = (const float*)d_in[18];
    const float* W2    = (const float*)d_in[19];
    const float* b2    = (const float*)d_in[20];
    const float* Wg    = (const float*)d_in[21];
    const float* bg    = (const float*)d_in[22];
    const float* lng_g = (const float*)d_in[23];
    const float* lng_b = (const float*)d_in[24];
    const float* Wt    = (const float*)d_in[25];
    const float* bt    = (const float*)d_in[26];

    _Float16* h1 = (_Float16*)d_ws;
    _Float16* h2 = h1 + (size_t)B_ROWS * RD;
    float* outG = (float*)d_out;
    float* outT = outG + (size_t)B_ROWS * COND;

    k1_roi_gemm_ln_gelu<<<dim3(8, 8), 256, 0, stream>>>(ra, b1, ln1g, ln1b, h1);
    k2_roi_linear<<<dim3(8, 8), 256, 0, stream>>>(h1, W2, b2, h2);
    k3_head_gemm<<<dim3(102, 8), 256, 0, stream>>>(h2, Wg, Wt, bg, bt, outG, outT);
    k4_row_ln<<<1024, 256, 0, stream>>>(outG, lng_g, lng_b);
}
